// MultiHeadAttention_84636625535558
// MI455X (gfx1250) — compile-verified, matrix operand order checked by reading, not run on hardware
//
#include <hip/hip_runtime.h>
#include <hip/hip_bf16.h>

#define DM     1024
#define DM3    3072
#define SEQ    2048
#define BATCH  2
#define NHEAD  16
#define DK     64
#define ROWS   (BATCH * SEQ)          // 4096
#define SCALE_F 0.125f                 // 1/sqrt(64)
#define MASK_FILL_F (-10000.0f)

typedef __attribute__((ext_vector_type(16))) _Float16 v16h;
typedef __attribute__((ext_vector_type(8)))  _Float16 v8h;
typedef __attribute__((ext_vector_type(8)))  float    v8f;

union F16x16 { v16h v; v8h h[2]; };

// Per-lane WMMA f16 operand fragment load (A or B), 16x32 tile, row-major src.
// Lane L holds row (L&15); K-chunks {e0..e0+7} and {e0+16..e0+23}, e0 = (L>>4)*8.
// Two 16-byte loads -> global_load_b128.
__device__ __forceinline__ v16h load_frag(const _Float16* tile, int ld, int lane) {
    int r  = lane & 15;
    int e0 = (lane >> 4) << 3;
    F16x16 u;
    u.h[0] = *(const v8h*)(tile + r * ld + e0);
    u.h[1] = *(const v8h*)(tile + r * ld + e0 + 16);
    return u.v;
}

__device__ __forceinline__ v8f wmma32(v16h a, v16h b, v8f c) {
    return __builtin_amdgcn_wmma_f32_16x16x32_f16(false, a, false, b, (short)0, c, false, false);
}

// ---------------- prep kernels ----------------
__global__ __launch_bounds__(256) void cvt_f16_kernel(const float* __restrict__ in,
                                                      _Float16* __restrict__ out, int n) {
    int i = blockIdx.x * blockDim.x + threadIdx.x;
    if (i < n) out[i] = (_Float16)in[i];
}

// in: rows x cols (f32) -> out: cols x rows (f16), N-major weights
__global__ __launch_bounds__(256) void transpose_cvt_kernel(const float* __restrict__ in,
                                                            _Float16* __restrict__ out,
                                                            int rows, int cols) {
    int i = blockIdx.x * blockDim.x + threadIdx.x;
    if (i < rows * cols) {
        int r = i / cols, c = i % cols;
        out[c * rows + r] = (_Float16)in[i];
    }
}

// ---------------- QKV projection ----------------
// C(4096x3072) = xh(4096x1024) @ Wqkv + bqkv ; scatter into Qh/Kh row-major, Vt transposed.
__global__ __launch_bounds__(256) void qkv_kernel(const _Float16* __restrict__ xh,
                                                  const _Float16* __restrict__ WqkvT,
                                                  const float* __restrict__ bqkv,
                                                  _Float16* __restrict__ Qh,
                                                  _Float16* __restrict__ Kh,
                                                  _Float16* __restrict__ Vt) {
    const int wave = threadIdx.x >> 5;
    const int lane = threadIdx.x & 31;
    const int tilesN = DM3 / 64;                 // 48
    const int bidN = blockIdx.x % tilesN;
    const int bidM = blockIdx.x / tilesN;
    const int rowBase = bidM * 128 + wave * 16;
    const int colBase = bidN * 64;

    v8f acc[4] = {};
    for (int k = 0; k < DM; k += 32) {
        v16h a = load_frag(xh + rowBase * DM + k, DM, lane);
#pragma unroll
        for (int nt = 0; nt < 4; ++nt) {
            v16h b = load_frag(WqkvT + (colBase + nt * 16) * DM + k, DM, lane);
            acc[nt] = wmma32(a, b, acc[nt]);
        }
    }

    const int hl = lane >> 4, n = lane & 15;
#pragma unroll
    for (int nt = 0; nt < 4; ++nt) {
        int col  = colBase + nt * 16 + n;
        int part = col / DM;                     // 0=Q 1=K 2=V
        int cm   = col % DM;
        int head = cm / DK, d = cm % DK;
        float bias = bqkv[col];
#pragma unroll
        for (int i = 0; i < 8; ++i) {
            int row = rowBase + i + 8 * hl;      // [0,4096)
            int bb = row / SEQ, s = row % SEQ;
            _Float16 val = (_Float16)(acc[nt][i] + bias);
            size_t hb = (size_t)(bb * NHEAD + head);
            if (part == 0)      Qh[(hb * SEQ + s) * DK + d] = val;
            else if (part == 1) Kh[(hb * SEQ + s) * DK + d] = val;
            else                Vt[(hb * DK + d) * SEQ + s] = val;
        }
    }
}

// ---------------- flash attention ----------------
// One wave per 16-query tile; online softmax over 32-key blocks.
__global__ __launch_bounds__(256) void attn_kernel(const _Float16* __restrict__ Qh,
                                                   const _Float16* __restrict__ Kh,
                                                   const _Float16* __restrict__ Vt,
                                                   const int* __restrict__ mask,
                                                   _Float16* __restrict__ ctx) {
    // per-wave 16x32 prob tile, row stride 40 halfs (80B -> 16B-aligned chunks)
    __shared__ __align__(16) _Float16 lds_p[8][16 * 40];

    const int wave = threadIdx.x >> 5;
    const int lane = threadIdx.x & 31;
    int qt = blockIdx.x * 8 + wave;              // 4096 q-tiles total
    int t = qt;
    const int qi   = (t % (SEQ / 16)) * 16; t /= (SEQ / 16);
    const int head = t % NHEAD;             t /= NHEAD;
    const int bb   = t;

    const _Float16* Qbase = Qh + ((size_t)(bb * NHEAD + head) * SEQ + qi) * DK;
    const _Float16* Kbase = Kh + (size_t)(bb * NHEAD + head) * SEQ * DK;
    const _Float16* Vbase = Vt + (size_t)(bb * NHEAD + head) * DK * SEQ;

    const v16h qf0 = load_frag(Qbase + 0,  DK, lane);   // d 0..31
    const v16h qf1 = load_frag(Qbase + 32, DK, lane);   // d 32..63

    const int hl = lane >> 4, n = lane & 15;
    const int e0 = hl * 8;
    _Float16* pl = &lds_p[wave][0];

    float rowmax[8], rowsum[8];
    v8f o[4] = {};
#pragma unroll
    for (int i = 0; i < 8; ++i) { rowmax[i] = -1e30f; rowsum[i] = 0.0f; }

    for (int kb = 0; kb < SEQ; kb += 32) {
        // ---- scores: two 16x16 tiles over DK=64 (2 WMMA each) ----
        v8f sc[2];
#pragma unroll
        for (int nt = 0; nt < 2; ++nt) {
            const _Float16* Kt = Kbase + (size_t)(kb + nt * 16) * DK;
            v16h b0 = load_frag(Kt + 0,  DK, lane);
            v16h b1 = load_frag(Kt + 32, DK, lane);
            v8f tacc = {};
            tacc = wmma32(qf0, b0, tacc);
            tacc = wmma32(qf1, b1, tacc);
            sc[nt] = tacc;
        }

        // ---- scale + mask (column per lane) ----
        const int m0 = mask[bb * SEQ + kb + n];
        const int m1 = mask[bb * SEQ + kb + 16 + n];
        float s0[8], s1[8];
#pragma unroll
        for (int i = 0; i < 8; ++i) {
            s0[i] = m0 ? sc[0][i] * SCALE_F : MASK_FILL_F;
            s1[i] = m1 ? sc[1][i] * SCALE_F : MASK_FILL_F;
        }

        // ---- online softmax: row max/sum across 16-lane half-groups ----
        float tm[8];
#pragma unroll
        for (int i = 0; i < 8; ++i) tm[i] = fmaxf(s0[i], s1[i]);
        for (int off = 1; off < 16; off <<= 1)
#pragma unroll
            for (int i = 0; i < 8; ++i) tm[i] = fmaxf(tm[i], __shfl_xor(tm[i], off, 32));

        float alpha[8], p0[8], p1[8], ps[8];
#pragma unroll
        for (int i = 0; i < 8; ++i) {
            float mnew = fmaxf(rowmax[i], tm[i]);
            alpha[i]  = __expf(rowmax[i] - mnew);
            rowmax[i] = mnew;
            p0[i] = __expf(s0[i] - mnew);
            p1[i] = __expf(s1[i] - mnew);
            ps[i] = p0[i] + p1[i];
        }
        for (int off = 1; off < 16; off <<= 1)
#pragma unroll
            for (int i = 0; i < 8; ++i) ps[i] += __shfl_xor(ps[i], off, 32);
#pragma unroll
        for (int i = 0; i < 8; ++i) rowsum[i] = rowsum[i] * alpha[i] + ps[i];
#pragma unroll
        for (int dt = 0; dt < 4; ++dt)
#pragma unroll
            for (int i = 0; i < 8; ++i) o[dt][i] *= alpha[i];

        // ---- re-layout P: C-layout -> A-fragment via LDS ----
#pragma unroll
        for (int i = 0; i < 8; ++i) {
            int row = i + 8 * hl;
            pl[row * 40 + n]      = (_Float16)p0[i];
            pl[row * 40 + 16 + n] = (_Float16)p1[i];
        }
        F16x16 u;
        u.h[0] = *(const v8h*)(pl + n * 40 + e0);
        u.h[1] = *(const v8h*)(pl + n * 40 + e0 + 16);
        v16h pf = u.v;

        // ---- context: O += P(16x32) @ V(32x16) for 4 d-tiles ----
#pragma unroll
        for (int dt = 0; dt < 4; ++dt) {
            F16x16 w;
            const _Float16* vrow = Vbase + (size_t)(dt * 16 + n) * SEQ + kb;
            w.h[0] = *(const v8h*)(vrow + e0);
            w.h[1] = *(const v8h*)(vrow + e0 + 16);
            o[dt] = wmma32(pf, w.v, o[dt]);
        }
    }

    // ---- epilogue: normalize, store ctx (f16) ----
#pragma unroll
    for (int dt = 0; dt < 4; ++dt)
#pragma unroll
        for (int i = 0; i < 8; ++i) {
            int row = i + 8 * hl;
            int s = qi + row;
            float val = o[dt][i] / rowsum[i];
            ctx[((size_t)(bb * SEQ + s)) * DM + head * DK + dt * 16 + n] = (_Float16)val;
        }
}

// ---------------- output projection ----------------
__global__ __launch_bounds__(256) void out_proj_kernel(const _Float16* __restrict__ ctx,
                                                       const _Float16* __restrict__ WoutT,
                                                       const float* __restrict__ bout,
                                                       float* __restrict__ out) {
    const int wave = threadIdx.x >> 5;
    const int lane = threadIdx.x & 31;
    const int tilesN = DM / 64;                  // 16
    const int bidN = blockIdx.x % tilesN;
    const int bidM = blockIdx.x / tilesN;
    const int rowBase = bidM * 128 + wave * 16;
    const int colBase = bidN * 64;

    v8f acc[4] = {};
    for (int k = 0; k < DM; k += 32) {
        v16h a = load_frag(ctx + rowBase * DM + k, DM, lane);
#pragma unroll
        for (int nt = 0; nt < 4; ++nt) {
            v16h b = load_frag(WoutT + (colBase + nt * 16) * DM + k, DM, lane);
            acc[nt] = wmma32(a, b, acc[nt]);
        }
    }
    const int hl = lane >> 4, n = lane & 15;
#pragma unroll
    for (int nt = 0; nt < 4; ++nt) {
        int col = colBase + nt * 16 + n;
        float bias = bout[col];
#pragma unroll
        for (int i = 0; i < 8; ++i) {
            int row = rowBase + i + 8 * hl;
            out[(size_t)row * DM + col] = acc[nt][i] + bias;
        }
    }
}

extern "C" void kernel_launch(void* const* d_in, const int* in_sizes, int n_in,
                              void* d_out, int out_size, void* d_ws, size_t ws_size,
                              hipStream_t stream) {
    const float* x    = (const float*)d_in[0];   // (2,2048,1024)
    const int*   mask = (const int*)  d_in[1];   // (2,2048)
    const float* Wqkv = (const float*)d_in[2];   // (1024,3072)
    const float* bqkv = (const float*)d_in[3];   // (3072)
    const float* Wout = (const float*)d_in[4];   // (1024,1024)
    const float* bout = (const float*)d_in[5];   // (1024)
    float* out = (float*)d_out;                  // (2,2048,1024)

    char* ws = (char*)d_ws;
    _Float16* xh     = (_Float16*)ws;                 ws += (size_t)ROWS * DM  * 2; // 8 MB
    _Float16* WqkvT  = (_Float16*)ws;                 ws += (size_t)DM3  * DM  * 2; // 6 MB
    _Float16* WoutT  = (_Float16*)ws;                 ws += (size_t)DM   * DM  * 2; // 2 MB
    _Float16* Qh     = (_Float16*)ws;                 ws += (size_t)ROWS * DM  * 2 / NHEAD * NHEAD; // 8 MB
    _Float16* Kh     = (_Float16*)ws;                 ws += (size_t)ROWS * DM  * 2; // 8 MB (Qh advanced same)
    _Float16* Vt     = (_Float16*)ws;                 ws += (size_t)ROWS * DM  * 2; // 8 MB
    _Float16* ctx    = (_Float16*)ws;                 // 8 MB

    // prep: f16 conversions / weight transposes
    cvt_f16_kernel<<<(ROWS * DM + 255) / 256, 256, 0, stream>>>(x, xh, ROWS * DM);
    transpose_cvt_kernel<<<(DM * DM3 + 255) / 256, 256, 0, stream>>>(Wqkv, WqkvT, DM, DM3);
    transpose_cvt_kernel<<<(DM * DM + 255) / 256, 256, 0, stream>>>(Wout, WoutT, DM, DM);

    // QKV projection: grid = (4096/128) * (3072/64)
    qkv_kernel<<<(ROWS / 128) * (DM3 / 64), 256, 0, stream>>>(xh, WqkvT, bqkv, Qh, Kh, Vt);

    // attention: 4096 q-tiles, 8 per block
    attn_kernel<<<(BATCH * NHEAD * (SEQ / 16)) / 8, 256, 0, stream>>>(Qh, Kh, Vt, mask, ctx);

    // output projection: grid = (4096/128) * (1024/64)
    out_proj_kernel<<<(ROWS / 128) * (DM / 64), 256, 0, stream>>>(ctx, WoutT, bout, out);
}
